// BatchLinear_59356448030869
// MI455X (gfx1250) — compile-verified, matrix operand order checked by reading, not run on hardware
//
#include <hip/hip_runtime.h>

typedef __attribute__((ext_vector_type(16))) _Float16 v16h;
typedef __attribute__((ext_vector_type(8)))  _Float16 v8h;
typedef __attribute__((ext_vector_type(4)))  _Float16 v4h;
typedef __attribute__((ext_vector_type(8)))  float    v8f;

#define U_DIM 128
#define V_DIM 128
#define B_ROWS 32768
#define NEXP 4
#define IN_STR 512
#define OUT_STR 512
#define W_STR (8 * U_DIM * V_DIM)   // 131072 floats per expert
#define SX_STRIDE 520               // 512 halves + 8 halves pad (16B) for LDS bank spread

// ---------------------------------------------------------------------------
// Kernel 1: transpose weight segments and fold the 0.5 path coefficient in.
// wt layout (f16): [(e*8 + j)][v][u]  (v = output col / N, u = input row / K)
// so B-fragment loads (fixed N column, running K) are contiguous.
// ---------------------------------------------------------------------------
__global__ __launch_bounds__(256) void bl_wconvert(const float* __restrict__ w,
                                                   _Float16* __restrict__ wt) {
    int t = blockIdx.x * 256 + threadIdx.x;      // 0 .. 524287
    int v = t & 127;
    int u = (t >> 7) & 127;
    int j = (t >> 14) & 7;
    int e = t >> 17;
    float val = w[(((e * 8 + j) * 128 + u) << 7) + v];
    if (j >= 4) val *= 0.5f;                     // paths 4..7 carry alpha = 0.5
    wt[(((e * 8 + j) * 128 + v) << 7) + u] = (_Float16)val;
}

// ---------------------------------------------------------------------------
// Kernel 2: bucket rows by expert id (single block, LDS atomics).
// Row values are position-independent, so nondeterministic slot order is
// value-deterministic for the final output.
// ---------------------------------------------------------------------------
__global__ __launch_bounds__(1024) void bl_index(const int* __restrict__ wid,
                                                 int* __restrict__ idx,
                                                 int* __restrict__ cnt) {
    __shared__ int c[NEXP];
    int tid = threadIdx.x;
    if (tid < NEXP) c[tid] = 0;
    __syncthreads();
    for (int b = tid; b < B_ROWS; b += 1024) {
        int e = wid[b];
        int p = atomicAdd(&c[e], 1);
        idx[e * B_ROWS + p] = b;
    }
    __syncthreads();
    if (tid < NEXP) cnt[tid] = c[tid];
}

// ---------------------------------------------------------------------------
// Kernel 3: expert-uniform tiled GEMM with v_wmma_f32_16x16x32_f16.
// Block = 8 wave32s, 32 gathered rows, full 512 output columns.
// Wave (mhalf, kseg): 16 rows x 128 cols, K = 256 (segments kseg and
// (kseg+3)%4 of x against weight segments kseg and (kseg+3)%4 + 4).
// ---------------------------------------------------------------------------
__global__ __launch_bounds__(256) void bl_gemm(const float* __restrict__ x,
                                               const _Float16* __restrict__ wt,
                                               const int* __restrict__ idx,
                                               const int* __restrict__ cnt,
                                               float* __restrict__ out) {
    const int e   = blockIdx.y;
    const int n_e = cnt[e];
    const int base = blockIdx.x * 32;
    if (base >= n_e) return;                     // block-uniform early exit

    __shared__ _Float16 sx[32 * SX_STRIDE];      // 33,280 B

    const int tid = threadIdx.x;

    // ---- stage 32 gathered rows of x, f32 -> f16 ----
    for (int s = tid; s < 32 * 128; s += 256) {
        int r   = s >> 7;                        // local row 0..31
        int c4  = s & 127;                       // float4 column
        int slot = base + r;
        int b = (slot < n_e) ? idx[e * B_ROWS + slot]
                             : idx[e * B_ROWS + n_e - 1];  // pad with a valid row
        float4 f = *(const float4*)(x + (size_t)b * IN_STR + c4 * 4);
        v4h h;
        h[0] = (_Float16)f.x; h[1] = (_Float16)f.y;
        h[2] = (_Float16)f.z; h[3] = (_Float16)f.w;
        *(v4h*)(&sx[r * SX_STRIDE + c4 * 4]) = h;
    }
    __syncthreads();

    const int lane  = tid & 31;
    const int wave  = tid >> 5;
    const int kseg  = wave & 3;                  // output segment 0..3
    const int mhalf = wave >> 2;                 // 0/1 -> rows 0-15 / 16-31
    const int lrow  = lane & 15;
    const int kb    = (lane >> 4) << 3;          // ISA 16-bit fragment K split

    v8f acc[8] = {};

    #pragma unroll
    for (int part = 0; part < 2; ++part) {
        const int aseg = part ? ((kseg + 3) & 3) : kseg;   // x segment
        const int wj   = part ? (aseg + 4)       : kseg;   // weight segment
        const _Float16* __restrict__ wseg = wt + (((size_t)e * 8 + wj) << 14);
        const _Float16* __restrict__ arow =
            &sx[(mhalf * 16 + lrow) * SX_STRIDE + aseg * 128];
        #pragma unroll
        for (int kk = 0; kk < 128; kk += 32) {
            // A fragment: lane = M row, halves K[kb..kb+7] then K[16+kb..16+kb+7]
            v8h alo = *(const v8h*)(arow + kk + kb);
            v8h ahi = *(const v8h*)(arow + kk + kb + 16);
            v16h a = __builtin_shufflevector(alo, ahi,
                0, 1, 2, 3, 4, 5, 6, 7, 8, 9, 10, 11, 12, 13, 14, 15);
            #pragma unroll
            for (int n = 0; n < 8; ++n) {
                // B fragment: lane = N column, same K split; contiguous in wt
                const _Float16* bcol = wseg + (((n * 16 + lrow) << 7) + kk + kb);
                v8h blo = *(const v8h*)(bcol);
                v8h bhi = *(const v8h*)(bcol + 16);
                v16h bb = __builtin_shufflevector(blo, bhi,
                    0, 1, 2, 3, 4, 5, 6, 7, 8, 9, 10, 11, 12, 13, 14, 15);
                acc[n] = __builtin_amdgcn_wmma_f32_16x16x32_f16(
                    false, a, false, bb, (short)0, acc[n], false, false);
            }
        }
    }

    // ---- scatter D tiles: lane holds rows rowbase..rowbase+7, col = lane&15 ----
    const int rowbase = mhalf * 16 + ((lane >> 4) << 3);
    int gb[8];
    #pragma unroll
    for (int r = 0; r < 8; ++r) {
        int slot = base + rowbase + r;
        gb[r] = (slot < n_e) ? idx[e * B_ROWS + slot] : -1;
    }
    #pragma unroll
    for (int n = 0; n < 8; ++n) {
        #pragma unroll
        for (int r = 0; r < 8; ++r) {
            if (gb[r] >= 0)
                out[(size_t)gb[r] * OUT_STR + kseg * 128 + n * 16 + lrow] = acc[n][r];
        }
    }
}

// ---------------------------------------------------------------------------
extern "C" void kernel_launch(void* const* d_in, const int* in_sizes, int n_in,
                              void* d_out, int out_size, void* d_ws, size_t ws_size,
                              hipStream_t stream) {
    const float* x  = (const float*)d_in[0];
    const float* w  = (const float*)d_in[1];
    const int* wid  = (const int*)d_in[2];
    float* out      = (float*)d_out;

    char* ws = (char*)d_ws;
    _Float16* wt = (_Float16*)ws;                                   // 1 MB
    int* idx = (int*)(ws + (size_t)NEXP * W_STR * sizeof(_Float16)); // 512 KB
    int* cnt = idx + (size_t)NEXP * B_ROWS;                          // 16 B

    bl_wconvert<<<dim3((NEXP * W_STR) / 256), 256, 0, stream>>>(w, wt);
    bl_index<<<1, 1024, 0, stream>>>(wid, idx, cnt);
    bl_gemm<<<dim3(B_ROWS / 32, NEXP), 256, 0, stream>>>(x, wt, idx, cnt, out);
}
